// MultiStatGuidedDomainAdapter_8907762171930
// MI455X (gfx1250) — matrix operand order repaired and checked by reading, not compiled, not run on hardware
//
#include <hip/hip_runtime.h>
#include <hip/hip_bf16.h>
#include <math.h>

// ---------------------------------------------------------------------------
// MultiStatGuidedDomainAdapter for MI455X (gfx1250, wave32, WMMA)
//
// Heavy part: 3x [8192x8192] Gram matrices (K=512) -> v_wmma_f32_16x16x32_bf16.
// Per 256-thread block (8 waves, 2x4): stage a 128x256 panel pair into LDS via
// GLOBAL_LOAD_ASYNC_TO_LDS_B128 (ASYNCcnt), each wave does a 64x64 region with
// 4x4 register blocking (16 WMMAs per K-step of 32).
// ---------------------------------------------------------------------------

typedef __attribute__((ext_vector_type(16))) __bf16 v16bf;
typedef __attribute__((ext_vector_type(8)))  float  v8f;

#define N_ROWS 8192
#define D_FEAT 512
#define EPS_F  1e-8f

#define BM_ROWS 128      // A panel rows per block
#define BN_ROWS 256      // B panel rows per block
#define KSTEP   32
#define LPAD    40       // padded LDS row length (bf16 elems) = 80B, 16B aligned

// workspace layout (float indices)
#define WS_SS      0
#define WS_TT      1
#define WS_ST      2
#define WS_LOGP    3
#define WS_LOG1MP  4
#define WS_AW      5
#define WS_SMOM    32
#define WS_TMOM    (WS_SMOM + 2560)
#define WS_SNORM   (WS_TMOM + 2560)
#define WS_TNORM   (WS_SNORM + 8192)
#define WS_BF16_BYTE_OFF ((size_t)(WS_TNORM + 8192) * 4)

// ---------------------------------------------------------------------------
__device__ __forceinline__ unsigned lds_off32(const void* p) {
    // LDS aperture addresses map to LDS by truncation to 32 bits (ISA 10.2)
    return (unsigned)(uintptr_t)p;
}

__device__ __forceinline__ void async_b128_to_lds(unsigned lds_off, const void* gptr) {
    asm volatile("global_load_async_to_lds_b128 %0, %1, off"
                 :: "v"(lds_off), "v"((unsigned long long)(uintptr_t)gptr)
                 : "memory");
}

__device__ __forceinline__ void wait_asynccnt0() {
    asm volatile("s_wait_asynccnt 0x0" ::: "memory");
}

// ---------------------------------------------------------------------------
__global__ void zero_accum_kernel(float* ws) {
    if (threadIdx.x < 32) ws[threadIdx.x] = 0.0f;
}

__global__ void convert_bf16_kernel(const float* __restrict__ in,
                                    __bf16* __restrict__ out, int n) {
    int i = blockIdx.x * blockDim.x + threadIdx.x;
    if (i < n) out[i] = (__bf16)in[i];
}

__global__ void rownorm_kernel(const float* __restrict__ x, float* __restrict__ norms) {
    __shared__ float red[256];
    const int row = blockIdx.x;
    const int tid = threadIdx.x;
    float a = x[row * D_FEAT + tid];
    float b = x[row * D_FEAT + tid + 256];
    red[tid] = a * a + b * b;
    __syncthreads();
    for (int s = 128; s > 0; s >>= 1) {
        if (tid < s) red[tid] += red[tid + s];
        __syncthreads();
    }
    if (tid == 0) norms[row] = red[0];
}

// per-feature moments: mean, unbiased var, kurtosis, skew, energy
__global__ void moments_kernel(const float* __restrict__ x, float* __restrict__ mom) {
    const int c   = threadIdx.x & 31;
    const int r   = threadIdx.x >> 5;
    const int col = blockIdx.x * 32 + c;

    __shared__ float sh1[8][32];
    __shared__ float sh2[8][32];
    __shared__ float smean[32], sinv[32];

    float s1 = 0.f, s2 = 0.f;
    for (int n = r; n < N_ROWS; n += 8) {
        float v = x[n * D_FEAT + col];
        s1 += v; s2 += v * v;
    }
    sh1[r][c] = s1; sh2[r][c] = s2;
    __syncthreads();
    if (r == 0) {
        float t1 = 0.f, t2 = 0.f;
        for (int i = 0; i < 8; ++i) { t1 += sh1[i][c]; t2 += sh2[i][c]; }
        float mean = t1 / (float)N_ROWS;
        float var  = (t2 - (float)N_ROWS * mean * mean) / (float)(N_ROWS - 1);
        var = fmaxf(var, 0.f);
        mom[0 * D_FEAT + col] = mean;
        mom[1 * D_FEAT + col] = var;
        mom[4 * D_FEAT + col] = t2 / (float)N_ROWS;
        smean[c] = mean;
        sinv[c]  = 1.0f / (sqrtf(var) + EPS_F);
    }
    __syncthreads();
    const float mean = smean[c], inv = sinv[c];
    float s3 = 0.f, s4 = 0.f;
    for (int n = r; n < N_ROWS; n += 8) {
        float z  = (x[n * D_FEAT + col] - mean) * inv;
        float z2 = z * z;
        s3 += z2 * z;
        s4 += z2 * z2;
    }
    sh1[r][c] = s3; sh2[r][c] = s4;
    __syncthreads();
    if (r == 0) {
        float t3 = 0.f, t4 = 0.f;
        for (int i = 0; i < 8; ++i) { t3 += sh1[i][c]; t4 += sh2[i][c]; }
        mom[2 * D_FEAT + col] = t4 / (float)N_ROWS - 3.0f;
        mom[3 * D_FEAT + col] = t3 / (float)N_ROWS;
    }
}

// stat diffs + adaptive-weight MLP 6->32->16->1
__global__ void statdiff_weight_kernel(const float* __restrict__ smom,
                                       const float* __restrict__ tmom,
                                       const float* __restrict__ wW1, const float* __restrict__ wb1,
                                       const float* __restrict__ wW2, const float* __restrict__ wb2,
                                       const float* __restrict__ wW3, const float* __restrict__ wb3,
                                       float* __restrict__ ws) {
    __shared__ float red[256];
    __shared__ float diffs[8];
    const int tid = threadIdx.x;
    for (int k = 0; k < 5; ++k) {
        float a = fabsf(smom[k * D_FEAT + tid]       - tmom[k * D_FEAT + tid]) +
                  fabsf(smom[k * D_FEAT + tid + 256] - tmom[k * D_FEAT + tid + 256]);
        red[tid] = a;
        __syncthreads();
        for (int s = 128; s > 0; s >>= 1) {
            if (tid < s) red[tid] += red[tid + s];
            __syncthreads();
        }
        if (tid == 0) diffs[k] = red[0] / (float)D_FEAT;
        __syncthreads();
    }
    if (tid == 0) {
        diffs[5] = 0.0f;
        float h1[32];
        for (int j = 0; j < 32; ++j) {
            float acc = wb1[j];
            for (int i = 0; i < 6; ++i) acc += diffs[i] * wW1[i * 32 + j];
            h1[j] = fmaxf(acc, 0.f);
        }
        float h2[16];
        for (int j = 0; j < 16; ++j) {
            float acc = wb2[j];
            for (int i = 0; i < 32; ++i) acc += h1[i] * wW2[i * 16 + j];
            h2[j] = fmaxf(acc, 0.f);
        }
        float z = wb3[0];
        for (int i = 0; i < 16; ++i) z += h2[i] * wW3[i];
        ws[WS_AW] = 1.0f / (1.0f + __expf(-z));
    }
}

// ---------------------------------------------------------------------------
// MMD Gram kernel: LDS-staged, async-copy fed, WMMA 4x4 register blocked.
// Block = 8 waves (2x4) -> 128x256 output region; wave -> 64x64.
// A fragment (16-bit 16x32): lane l holds row M=l%16, 16 contiguous K from kb0.
// B fragment (32x16, B=Y^T): lane l holds col N=l%16, 16 contiguous K of Y row.
// C/D: VGPR r, lane l -> M = r + 8*(l/16), N = l%16.
// ---------------------------------------------------------------------------
__global__ void __launch_bounds__(256)
mmd_wmma_kernel(const __bf16* __restrict__ sb, const __bf16* __restrict__ tb,
                const float* __restrict__ snorm, const float* __restrict__ tnorm,
                float* __restrict__ sums) {
    __shared__ __attribute__((aligned(16))) __bf16 Abuf[BM_ROWS][LPAD];
    __shared__ __attribute__((aligned(16))) __bf16 Bbuf[BN_ROWS][LPAD];

    const int tid  = threadIdx.x;
    const int lane = tid & 31;
    const int wave = tid >> 5;
    const int wm   = wave >> 2;          // 0..1
    const int wn   = wave & 3;           // 0..3

    const int GM   = N_ROWS / BM_ROWS;   // 64
    const int GN   = N_ROWS / BN_ROWS;   // 32
    const int prod = blockIdx.x / (GM * GN);          // 0:ss 1:tt 2:st
    const int rem  = blockIdx.x - prod * GM * GN;
    const int bm   = rem / GN;
    const int bn   = rem - bm * GN;

    const __bf16* Am = (prod == 1) ? tb : sb;
    const __bf16* Bm = (prod == 0) ? sb : tb;
    const float*  an = (prod == 1) ? tnorm : snorm;
    const float*  bv = (prod == 0) ? snorm : tnorm;

    const __bf16* Agbase = Am + (size_t)(bm * BM_ROWS) * D_FEAT;
    const __bf16* Bgbase = Bm + (size_t)(bn * BN_ROWS) * D_FEAT;

    const int rsel  = lane & 15;
    const int kb0   = (lane < 16) ? 0 : 16;
    const int koffB = (lane < 16) ? 0 : 16;

    v8f acc[4][4];
#pragma unroll
    for (int mi = 0; mi < 4; ++mi)
#pragma unroll
        for (int ni = 0; ni < 4; ++ni) acc[mi][ni] = (v8f){};

    for (int kt = 0; kt < D_FEAT; kt += KSTEP) {
        // ---- async stage: A panel 128x32 (512 b128 chunks), B panel 256x32 (1024)
#pragma unroll
        for (int j = 0; j < 2; ++j) {                    // A: 2 chunks / thread
            int q   = tid + j * 256;
            int row = q >> 2, kc = q & 3;
            async_b128_to_lds(lds_off32(&Abuf[row][kc * 8]),
                              Agbase + (size_t)row * D_FEAT + kt + kc * 8);
        }
#pragma unroll
        for (int j = 0; j < 4; ++j) {                    // B: 4 chunks / thread
            int q   = tid + j * 256;
            int row = q >> 2, kc = q & 3;
            async_b128_to_lds(lds_off32(&Bbuf[row][kc * 8]),
                              Bgbase + (size_t)row * D_FEAT + kt + kc * 8);
        }
        wait_asynccnt0();
        __syncthreads();

        // ---- fragments from LDS
        v16bf af[4], bfr[4];
#pragma unroll
        for (int mi = 0; mi < 4; ++mi) {
            const __bf16* a0 = &Abuf[wm * 64 + mi * 16 + rsel][kb0];
#pragma unroll
            for (int e = 0; e < 8; ++e) { af[mi][e] = a0[e]; af[mi][e + 8] = a0[8 + e]; }
        }
#pragma unroll
        for (int ni = 0; ni < 4; ++ni) {
            const __bf16* b0 = &Bbuf[wn * 64 + ni * 16 + rsel][koffB];
#pragma unroll
            for (int e = 0; e < 16; ++e) bfr[ni][e] = b0[e];
        }

        // ---- 16 WMMAs
#pragma unroll
        for (int mi = 0; mi < 4; ++mi)
#pragma unroll
            for (int ni = 0; ni < 4; ++ni)
                acc[mi][ni] = __builtin_amdgcn_wmma_f32_16x16x32_bf16(
                    false, af[mi], false, bfr[ni], (short)0, acc[mi][ni], false, false);

        __syncthreads();    // buffers reused next K-step
    }

    // ---- epilogue: d2 = |x_i|^2 + |y_j|^2 - 2*dot ; accumulate exp(-d2/2)
    const int mrow0 = bm * BM_ROWS + wm * 64 + ((lane >> 4) << 3);
    const int jcol0 = bn * BN_ROWS + wn * 64 + rsel;
    float local = 0.f;
#pragma unroll
    for (int mi = 0; mi < 4; ++mi) {
#pragma unroll
        for (int ni = 0; ni < 4; ++ni) {
            const float bnj = bv[jcol0 + ni * 16];
#pragma unroll
            for (int r = 0; r < 8; ++r) {
                float d2 = an[mrow0 + mi * 16 + r] + bnj - 2.0f * acc[mi][ni][r];
                local += __expf(-0.5f * d2);
            }
        }
    }
    for (int off = 16; off > 0; off >>= 1)
        local += __shfl_xor(local, off, 32);
    if (lane == 0) atomicAdd(&sums[prod], local);
}

// ---------------------------------------------------------------------------
// Domain discriminator MLP (512->256->128->1) + clamped-log accumulation.
// ---------------------------------------------------------------------------
__global__ void __launch_bounds__(256)
disc_kernel(const float* __restrict__ x,
            const float* __restrict__ W1, const float* __restrict__ b1,
            const float* __restrict__ W2, const float* __restrict__ b2,
            const float* __restrict__ W3, const float* __restrict__ b3,
            float* __restrict__ accum, int label_one) {
    __shared__ float xrow[512];
    __shared__ float h1[256];
    __shared__ float h2[128];
    __shared__ float red[4];
    const int row = blockIdx.x;
    const int tid = threadIdx.x;

    xrow[tid]       = x[(size_t)row * D_FEAT + tid];
    xrow[tid + 256] = x[(size_t)row * D_FEAT + tid + 256];
    __syncthreads();
    {
        float acc = b1[tid];
        for (int k = 0; k < 512; ++k) acc += xrow[k] * W1[k * 256 + tid];
        h1[tid] = fmaxf(acc, 0.f);
    }
    __syncthreads();
    if (tid < 128) {
        float acc = b2[tid];
        for (int k = 0; k < 256; ++k) acc += h1[k] * W2[k * 128 + tid];
        h2[tid] = fmaxf(acc, 0.f);
    }
    __syncthreads();
    if (tid < 128) {
        float p = h2[tid] * W3[tid];
        for (int off = 16; off > 0; off >>= 1) p += __shfl_xor(p, off, 32);
        if ((tid & 31) == 0) red[tid >> 5] = p;
    }
    __syncthreads();
    if (tid == 0) {
        float z    = red[0] + red[1] + red[2] + red[3] + b3[0];
        float pred = 1.0f / (1.0f + __expf(-z));
        float l    = label_one ? logf(pred) : logf(1.0f - pred);
        l = fmaxf(l, -100.0f);
        atomicAdd(accum, l);
    }
}

__global__ void finalize_kernel(const float* __restrict__ ws, float* __restrict__ out) {
    if (threadIdx.x == 0 && blockIdx.x == 0) {
        const float NN  = (float)N_ROWS * (float)N_ROWS;
        float mmd       = ws[WS_SS] / NN + ws[WS_TT] / NN - 2.0f * ws[WS_ST] / NN;
        float bce_src   = -ws[WS_LOGP]   / (float)N_ROWS;
        float bce_tgt   = -ws[WS_LOG1MP] / (float)N_ROWS;
        float adv       = bce_src + bce_tgt;
        float aw        = ws[WS_AW];
        out[0] = aw * mmd + (1.0f - aw) * adv;
        out[1] = aw;
    }
}

// ---------------------------------------------------------------------------
extern "C" void kernel_launch(void* const* d_in, const int* in_sizes, int n_in,
                              void* d_out, int out_size, void* d_ws, size_t ws_size,
                              hipStream_t stream) {
    const float* src = (const float*)d_in[0];
    const float* tgt = (const float*)d_in[1];
    const float* dW1 = (const float*)d_in[2];
    const float* db1 = (const float*)d_in[3];
    const float* dW2 = (const float*)d_in[4];
    const float* db2 = (const float*)d_in[5];
    const float* dW3 = (const float*)d_in[6];
    const float* db3 = (const float*)d_in[7];
    const float* wW1 = (const float*)d_in[8];
    const float* wb1 = (const float*)d_in[9];
    const float* wW2 = (const float*)d_in[10];
    const float* wb2 = (const float*)d_in[11];
    const float* wW3 = (const float*)d_in[12];
    const float* wb3 = (const float*)d_in[13];

    float*  ws = (float*)d_ws;
    __bf16* sb = (__bf16*)((char*)d_ws + WS_BF16_BYTE_OFF);
    __bf16* tb = sb + (size_t)N_ROWS * D_FEAT;

    const int NE = N_ROWS * D_FEAT;

    zero_accum_kernel<<<1, 32, 0, stream>>>(ws);

    convert_bf16_kernel<<<NE / 256, 256, 0, stream>>>(src, sb, NE);
    convert_bf16_kernel<<<NE / 256, 256, 0, stream>>>(tgt, tb, NE);

    rownorm_kernel<<<N_ROWS, 256, 0, stream>>>(src, ws + WS_SNORM);
    rownorm_kernel<<<N_ROWS, 256, 0, stream>>>(tgt, ws + WS_TNORM);

    moments_kernel<<<D_FEAT / 32, 256, 0, stream>>>(src, ws + WS_SMOM);
    moments_kernel<<<D_FEAT / 32, 256, 0, stream>>>(tgt, ws + WS_TMOM);

    statdiff_weight_kernel<<<1, 256, 0, stream>>>(ws + WS_SMOM, ws + WS_TMOM,
                                                  wW1, wb1, wW2, wb2, wW3, wb3, ws);

    // 3 products x (64 x 32) 128x256 block-tiles -> exact grid, EXEC all-1s
    const int nblocks = 3 * (N_ROWS / BM_ROWS) * (N_ROWS / BN_ROWS);
    mmd_wmma_kernel<<<nblocks, 256, 0, stream>>>(sb, tb,
                                                 ws + WS_SNORM, ws + WS_TNORM, ws);

    disc_kernel<<<N_ROWS, 256, 0, stream>>>(src, dW1, db1, dW2, db2, dW3, db3,
                                            ws + WS_LOGP, 1);
    disc_kernel<<<N_ROWS, 256, 0, stream>>>(tgt, dW1, db1, dW2, db2, dW3, db3,
                                            ws + WS_LOG1MP, 0);

    finalize_kernel<<<1, 1, 0, stream>>>(ws, (float*)d_out);
}
